// GraphAttentionBias_29154238005998
// MI455X (gfx1250) — matrix operand order repaired and checked by reading, not compiled, not run on hardware
//
#include <hip/hip_runtime.h>

typedef __attribute__((ext_vector_type(16))) _Float16 v16h;
typedef __attribute__((ext_vector_type(8)))  float    v8f;
typedef __attribute__((ext_vector_type(4)))  float    v4f;

#define Bc   32
#define Nc   1024
#define Ec   16384
#define DSc  8
#define Hc   32
#define EPSc 1e-12f

// ---------------------------------------------------------------------------
// Kernel 1: fill 128 MiB output with default_bias. Pure-bandwidth: float4
// non-temporal stores so the giant stream doesn't thrash L2.
// ---------------------------------------------------------------------------
__global__ void __launch_bounds__(256)
gab_fill_kernel(float* __restrict__ out, const float* __restrict__ default_bias) {
    size_t i = (size_t)blockIdx.x * blockDim.x + threadIdx.x;
    float v = default_bias[0];
    v4f f = { v, v, v, v };
    __builtin_nontemporal_store(f, ((v4f*)out) + i);
}

// ---------------------------------------------------------------------------
// Kernel 2: per-edge MLP via WMMA + symmetric scatter.
// One wave = 16 (batch, edge) pairs:
//   A (16x32 f16): rows = edges, cols = [dist, rest, dist-rest, te0..te7, 0...]
//   B (32x16 f16): w1 (zero-padded K); two WMMAs cover the 32 hidden units
//   C (16x16 f32): b1 broadcast
// Final 32->1 dot: relu(D)*w2[n] per lane, then shfl_xor reduction over the
// 16-lane halves (matches D-layout row split), then scatter from lanes 0/16.
// ---------------------------------------------------------------------------
__global__ void __launch_bounds__(256)
gab_score_scatter_kernel(const float* __restrict__ xyz,
                         const int*   __restrict__ edge_index,
                         const int*   __restrict__ edge_type,
                         const float* __restrict__ rest_len,
                         const float* __restrict__ type_emb,
                         const float* __restrict__ w1,
                         const float* __restrict__ b1,
                         const float* __restrict__ w2,
                         const float* __restrict__ b2,
                         float* __restrict__ out) {
    const int wave = blockIdx.x * (blockDim.x >> 5) + (threadIdx.x >> 5);
    const int lane = threadIdx.x & 31;
    const int hi   = lane >> 4;      // lane half: 0 -> rows 0-7, 1 -> rows 8-15
    const int n    = lane & 15;      // hidden column (B/C/D) and edge row (A)

    const int tilesPerBatch = Ec / 16;               // 1024
    const int b  = wave / tilesPerBatch;
    const int e0 = (wave % tilesPerBatch) * 16;

    // --- per-lane edge features (edge row n of this tile) ---
    const int   e    = e0 + n;
    const int   srcI = edge_index[e];
    const int   dstI = edge_index[Ec + e];
    const int   et   = edge_type[e];
    const float rest = rest_len[e];
    const float* ps  = xyz + ((size_t)b * Nc + srcI) * 3;
    const float* pd  = xyz + ((size_t)b * Nc + dstI) * 3;
    const float dx = pd[0] - ps[0];
    const float dy = pd[1] - ps[1];
    const float dz = pd[2] - ps[2];
    const float dist = sqrtf(dx * dx + dy * dy + dz * dz + EPSc);
    const float* te  = type_emb + et * DSc;

    // --- A fragment: 16x32, contiguous k = 16*half + elem ---
    // lanes 0-15 : elems 0-15 = K0..15 (pad -> 0)
    // lanes 16-31: elems 0-15 = K16..31 (all pad -> 0)
    v16h a;
#pragma unroll
    for (int i = 0; i < 16; ++i) a[i] = (_Float16)0.0f;
    if (hi == 0) {
        a[0] = (_Float16)dist;
        a[1] = (_Float16)rest;
        a[2] = (_Float16)(dist - rest);
        a[3] = (_Float16)te[0];
        a[4] = (_Float16)te[1];
        a[5] = (_Float16)te[2];
        a[6] = (_Float16)te[3];
        a[7] = (_Float16)te[4];
        a[8] = (_Float16)te[5];
        a[9] = (_Float16)te[6];
        a[10] = (_Float16)te[7];
    }

    // --- B fragments: 32x16, contiguous k = 16*half + elem ---
    // lanes 0-15 hold K=0..15 (elems 0..15), lanes 16-31 hold K=16..31 (all pad -> 0)
    v16h wb0, wb1;
#pragma unroll
    for (int i = 0; i < 16; ++i) { wb0[i] = (_Float16)0.0f; wb1[i] = (_Float16)0.0f; }
    if (hi == 0) {
#pragma unroll
        for (int k = 0; k < 11; ++k) {
            wb0[k] = (_Float16)w1[k * Hc + n];        // hidden 0..15
            wb1[k] = (_Float16)w1[k * Hc + n + 16];   // hidden 16..31
        }
    }

    // --- C fragments: b1 broadcast per hidden column ---
    const float bn0 = b1[n];
    const float bn1 = b1[n + 16];
    v8f c0, c1;
#pragma unroll
    for (int i = 0; i < 8; ++i) { c0[i] = bn0; c1[i] = bn1; }

    v8f h0 = __builtin_amdgcn_wmma_f32_16x16x32_f16(false, a, false, wb0,
                                                    (short)0, c0, false, false);
    v8f h1 = __builtin_amdgcn_wmma_f32_16x16x32_f16(false, a, false, wb1,
                                                    (short)0, c1, false, false);

    // --- relu + weighted partial dot, then 16-lane xor reduction ---
    const float w2n0 = w2[n];
    const float w2n1 = w2[n + 16];
    float p[8];
#pragma unroll
    for (int v = 0; v < 8; ++v)
        p[v] = fmaxf(h0[v], 0.0f) * w2n0 + fmaxf(h1[v], 0.0f) * w2n1;

#pragma unroll
    for (int mask = 1; mask < 16; mask <<= 1) {
#pragma unroll
        for (int v = 0; v < 8; ++v)
            p[v] += __shfl_xor(p[v], mask, 32);
    }

    // --- symmetric scatter: lane 0 -> rows 0-7, lane 16 -> rows 8-15 ---
    if (n == 0) {
        const float  b2v   = b2[0];
        const int    base  = e0 + hi * 8;
        const size_t obase = (size_t)b * Nc * Nc;
#pragma unroll
        for (int j = 0; j < 8; ++j) {
            const int   ee = base + j;
            const int   s  = edge_index[ee];
            const int   d  = edge_index[Ec + ee];
            const float sc = p[j] + b2v;
            out[obase + (size_t)s * Nc + d] = sc;
            out[obase + (size_t)d * Nc + s] = sc;
        }
    }
}

// ---------------------------------------------------------------------------
extern "C" void kernel_launch(void* const* d_in, const int* in_sizes, int n_in,
                              void* d_out, int out_size, void* d_ws, size_t ws_size,
                              hipStream_t stream) {
    const float* xyz          = (const float*)d_in[0];
    const int*   edge_index   = (const int*)  d_in[1];
    const int*   edge_type    = (const int*)  d_in[2];
    const float* rest_len     = (const float*)d_in[3];
    const float* type_emb     = (const float*)d_in[4];
    const float* w1           = (const float*)d_in[5];
    const float* b1           = (const float*)d_in[6];
    const float* w2           = (const float*)d_in[7];
    const float* b2           = (const float*)d_in[8];
    const float* default_bias = (const float*)d_in[9];
    float* out = (float*)d_out;

    // 1) Fill 32*1024*1024 floats with default_bias (float4 NT stores).
    const size_t n4 = (size_t)Bc * Nc * Nc / 4;   // 8,388,608
    gab_fill_kernel<<<(unsigned)(n4 / 256), 256, 0, stream>>>(out, default_bias);

    // 2) WMMA edge MLP + symmetric scatter.
    //    B*E/16 = 32768 waves -> 4096 blocks of 256 threads (8 waves/block).
    gab_score_scatter_kernel<<<4096, 256, 0, stream>>>(
        xyz, edge_index, edge_type, rest_len, type_emb, w1, b1, w2, b2, out);
}
